// VoxurfC_53841710023270
// MI455X (gfx1250) — matrix operand order repaired and checked by reading, not compiled, not run on hardware
//
#include <hip/hip_runtime.h>
#include <hip/hip_bf16.h>
#include <math.h>

typedef __attribute__((ext_vector_type(16))) _Float16 v16h;
typedef __attribute__((ext_vector_type(8)))  _Float16 v8h;
typedef __attribute__((ext_vector_type(8)))  float    v8f;

#define NRAYS   4096
#define NSTEPS  160
#define NPTS    (NRAYS*NSTEPS)     // 655360
#define GG      160
#define G3      (GG*GG*GG)         // 4,096,000
#define GM      80
#define NEARV   0.05f
#define VOXEL   (2.0f/160.0f)
#define DISTV   (0.5f*VOXEL)       // 0.00625
#define SVAL    80.0f
#define ACT_SHIFT (-9.21024037f)
#define MASK_THRES 1e-3f
#define FAST_THRES 1e-7f

// packed-weight layout (halves, per MLP)
#define WP_L0   0       // 3 ktiles * 8 ntiles * 512 = 12288
#define WP_L1   12288   // 4 * 8 * 512 = 16384
#define WP_L2   28672   // 4 * 1 * 512 = 2048
#define WP_MLP  30720

// ---------------- helpers ----------------

__device__ __forceinline__ void tsetup(float p, int Gx, int& i0, float& f) {
    float u = (p + 1.0f) * 0.5f * (float)(Gx - 1);
    u = fminf(fmaxf(u, 0.0f), (float)(Gx - 1));
    int i = (int)floorf(u);
    if (i > Gx - 2) i = Gx - 2;
    f = u - (float)i;
    i0 = i;
}

__device__ __forceinline__ float trilerp_ch(const float* __restrict__ g, int Gx,
                                            int x0, int y0, int z0,
                                            float fx, float fy, float fz) {
    long G2 = (long)Gx * Gx;
    const float* b = g + (long)x0 * G2 + (long)y0 * Gx + z0;
    float c000 = b[0],      c001 = b[1];
    float c010 = b[Gx],     c011 = b[Gx + 1];
    float c100 = b[G2],     c101 = b[G2 + 1];
    float c110 = b[G2 + Gx],c111 = b[G2 + Gx + 1];
    float c00 = c000 * (1.f - fz) + c001 * fz;
    float c01 = c010 * (1.f - fz) + c011 * fz;
    float c10 = c100 * (1.f - fz) + c101 * fz;
    float c11 = c110 * (1.f - fz) + c111 * fz;
    float c0 = c00 * (1.f - fy) + c01 * fy;
    float c1 = c10 * (1.f - fy) + c11 * fy;
    return c0 * (1.f - fx) + c1 * fx;
}

__device__ __forceinline__ float sigmoidf(float x) { return 1.0f / (1.0f + expf(-x)); }
__device__ __forceinline__ float softplusf(float x) {
    return fmaxf(x, 0.0f) + log1pf(expf(-fabsf(x)));
}

// ---------------- grid preprocessing ----------------

__global__ void gauss1d(const float* __restrict__ in, float* __restrict__ out,
                        int stride, int axisSel) {
    int id = blockIdx.x * blockDim.x + threadIdx.x;
    if (id >= G3) return;
    int z = id % GG; int y = (id / GG) % GG; int x = id / (GG * GG);
    int c = (axisSel == 0) ? x : ((axisSel == 1) ? y : z);
    const float w[5] = {0.0544886845f, 0.2442013420f, 0.4026199470f,
                        0.2442013420f, 0.0544886845f};
    float acc = 0.f;
#pragma unroll
    for (int o = -2; o <= 2; ++o) {
        int cc = c + o;
        if (cc >= 0 && cc < GG) acc += w[o + 2] * in[id + o * stride];
    }
    out[id] = acc;
}

__global__ void grad_kernel(const float* __restrict__ s, float* __restrict__ g) {
    int id = blockIdx.x * blockDim.x + threadIdx.x;
    if (id >= G3) return;
    int z = id % GG; int y = (id / GG) % GG; int x = id / (GG * GG);
    const float inv2v = 1.0f / (2.0f * VOXEL);
    float gx = (x > 0 && x < GG - 1) ? (s[id + GG * GG] - s[id - GG * GG]) * inv2v : 0.f;
    float gy = (y > 0 && y < GG - 1) ? (s[id + GG] - s[id - GG]) * inv2v : 0.f;
    float gz = (z > 0 && z < GG - 1) ? (s[id + 1] - s[id - 1]) * inv2v : 0.f;
    g[id] = gx; g[G3 + id] = gy; g[2 * G3 + id] = gz;
}

// pack fp32 weights into f16 WMMA B-fragment layout
__global__ void pack_weights(const float* __restrict__ ow0, const float* __restrict__ ow1,
                             const float* __restrict__ ow2, const float* __restrict__ ew0,
                             const float* __restrict__ ew1, const float* __restrict__ ew2,
                             _Float16* __restrict__ wp) {
    int g = blockIdx.x * blockDim.x + threadIdx.x;
    if (g >= 2 * WP_MLP) return;
    int m = g / WP_MLP;
    int r = g % WP_MLP;
    const float* w; int off, NT, Kreal, Nreal;
    if (r < WP_L1)      { off = r - WP_L0; NT = 8; Kreal = 75;  Nreal = 128; w = m ? ew0 : ow0; }
    else if (r < WP_L2) { off = r - WP_L1; NT = 8; Kreal = 128; Nreal = 128; w = m ? ew1 : ow1; }
    else                { off = r - WP_L2; NT = 1; Kreal = 128; Nreal = 3;   w = m ? ew2 : ow2; }
    int e    = off & 15;
    int lane = (off >> 4) & 31;
    int tile = off >> 9;
    int nt = tile % NT;
    int kt = tile / NT;
    int k = kt * 32 + ((lane & 16) ? 8 : 0) + (e & 7) + ((e & 8) ? 16 : 0);   // B fragment: table K order (8h+s, 16+8h+s-8)
    int n = nt * 16 + (lane & 15);
    float v = 0.f;
    if (k < Kreal && n < Nreal) v = w[k * Nreal + n];
    wp[g] = (_Float16)v;
}

// ---------------- per-point alpha ----------------

__global__ void alpha_kernel(const float* __restrict__ rays_o, const float* __restrict__ rays_d,
                             const float* __restrict__ viewdirs,
                             const float* __restrict__ mask_density,
                             const float* __restrict__ sdfs, const float* __restrict__ grad,
                             float* __restrict__ alpha_out, float* __restrict__ gpt) {
    int p = blockIdx.x * blockDim.x + threadIdx.x;
    if (p >= NPTS) return;
    int ray = p / NSTEPS, step = p % NSTEPS;
    float t = NEARV + DISTV * (float)step;
    float dx = rays_d[ray * 3 + 0], dy = rays_d[ray * 3 + 1], dz = rays_d[ray * 3 + 2];
    float px = rays_o[ray * 3 + 0] + dx * t;
    float py = rays_o[ray * 3 + 1] + dy * t;
    float pz = rays_o[ray * 3 + 2] + dz * t;

    // mask cache (80^3)
    int mx, my, mz; float mfx, mfy, mfz;
    tsetup(px, GM, mx, mfx); tsetup(py, GM, my, mfy); tsetup(pz, GM, mz, mfz);
    float dens = trilerp_ch(mask_density, GM, mx, my, mz, mfx, mfy, mfz);
    float mask_alpha = 1.0f - expf(-softplusf(dens + ACT_SHIFT) * DISTV);
    bool keep = mask_alpha >= MASK_THRES;

    int x0, y0, z0; float fx, fy, fz;
    tsetup(px, GG, x0, fx); tsetup(py, GG, y0, fy); tsetup(pz, GG, z0, fz);
    float sdf = trilerp_ch(sdfs, GG, x0, y0, z0, fx, fy, fz);
    float g0 = trilerp_ch(grad,          GG, x0, y0, z0, fx, fy, fz);
    float g1 = trilerp_ch(grad + G3,     GG, x0, y0, z0, fx, fy, fz);
    float g2 = trilerp_ch(grad + 2 * G3, GG, x0, y0, z0, fx, fy, fz);

    float vx = viewdirs[ray * 3 + 0], vy = viewdirs[ray * 3 + 1], vz = viewdirs[ray * 3 + 2];
    float tc = vx * g0 + vy * g1 + vz * g2;
    float ic = fminf(tc, 0.0f);                 // -relu(-tc)
    float e  = 0.5f * DISTV * ic;
    float prev = sigmoidf((sdf - e) * SVAL);
    float next = sigmoidf((sdf + e) * SVAL);
    float a = (prev - next + 1e-5f) / (prev + 1e-5f);
    a = fminf(fmaxf(a, 0.0f), 1.0f);
    alpha_out[p] = keep ? a : 0.0f;
    gpt[p * 3 + 0] = g0; gpt[p * 3 + 1] = g1; gpt[p * 3 + 2] = g2;
}

// ---------------- per-ray transmittance scan ----------------
// keep1 at step s depends only on alpha[0..s] -> single fused forward pass.

__global__ void weights_kernel(const float* __restrict__ alpha, float* __restrict__ weights,
                               float* __restrict__ out) {
    int ray = blockIdx.x * blockDim.x + threadIdx.x;
    if (ray >= NRAYS) return;
    float T1 = 1.f, T2 = 1.f, cum = 0.f;
    for (int s = 0; s < NSTEPS; ++s) {
        float a  = alpha[ray * NSTEPS + s];
        float w1 = a * T1; T1 *= (1.f - a);
        float a2 = (w1 > FAST_THRES) ? a : 0.f;
        float w2 = a2 * T2; T2 *= (1.f - a2);
        cum += w2;
        weights[ray * NSTEPS + s] = w2;
    }
    out[NRAYS * 3 + ray] = T2;          // alphainv_last
    out[NRAYS * 4 + ray] = 1.0f - cum;  // 1 - cum_weights
}

// ---------------- WMMA MLP ----------------
// 128 points per workgroup; 4 waves * 2 M-tiles; f16 WMMA 16x16x32.

#define PTS 128
#define FS  96   // padded K for layer 0

__device__ __forceinline__ v16h load_a_frag(const _Float16* ap) {
    v8h lo = *reinterpret_cast<const v8h*>(ap);        // K = kb + 0..7
    v8h hi = *reinterpret_cast<const v8h*>(ap + 16);   // K = kb + 16..23
    v16h a;
#pragma unroll
    for (int i = 0; i < 8; ++i) { a[i] = lo[i]; a[8 + i] = hi[i]; }
    return a;
}

__device__ __forceinline__ void wmma_layer(const _Float16* __restrict__ src, int srcStride,
                                           int KT, const _Float16* __restrict__ wp, int NT,
                                           const float* __restrict__ bias,
                                           _Float16* __restrict__ dst,
                                           int wave, int lane) {
    for (int mt = wave * 2; mt < wave * 2 + 2; ++mt) {
        int row = mt * 16 + (lane & 15);
        int kb  = (lane & 16) ? 8 : 0;
        for (int nt = 0; nt < NT; ++nt) {
            float bn = bias[nt * 16 + (lane & 15)];
            v8f c = {bn, bn, bn, bn, bn, bn, bn, bn};
            for (int kt = 0; kt < KT; ++kt) {
                v16h a = load_a_frag(src + row * srcStride + kt * 32 + kb);
                v16h b = *reinterpret_cast<const v16h*>(wp + ((kt * NT + nt) * 32 + lane) * 16);
                c = __builtin_amdgcn_wmma_f32_16x16x32_f16(false, a, false, b,
                                                           (short)0, c, false, false);
            }
            int n = nt * 16 + (lane & 15);
            int mbase = mt * 16 + ((lane >> 4) << 3);
#pragma unroll
            for (int r = 0; r < 8; ++r) {
                float v = c[r];
                dst[(mbase + r) * 128 + n] = (_Float16)(v > 0.f ? v : 0.f);  // ReLU
            }
        }
    }
}

__global__ __launch_bounds__(PTS)
void mlp_kernel(const float* __restrict__ rays_o, const float* __restrict__ rays_d,
                const float* __restrict__ viewdirs, const int* __restrict__ em_modes,
                const float* __restrict__ offc, const float* __restrict__ emoc_g,
                const _Float16* __restrict__ wpack,
                const float* __restrict__ off_b0, const float* __restrict__ off_b1,
                const float* __restrict__ off_b2, const float* __restrict__ emo_b0,
                const float* __restrict__ emo_b1, const float* __restrict__ emo_b2,
                const float* __restrict__ gpt, const float* __restrict__ weights,
                float* __restrict__ wrgb) {
    __shared__ _Float16 feat[PTS * FS];     // 24 KB
    __shared__ _Float16 hA[PTS * 128];      // 32 KB
    __shared__ _Float16 hB[PTS * 128];      // 32 KB
    __shared__ _Float16 emocol[PTS * 4];
    __shared__ float    outrgb[PTS * 3];
    __shared__ int      emf[PTS];

    const int tid  = threadIdx.x;
    const int base = blockIdx.x * PTS;
    const int wave = tid >> 5;
    const int lane = tid & 31;

    __builtin_prefetch(wpack, 0, 1);        // global_prefetch_b8: weights -> cache

    // ---- phase 1: per-point features into A-fragment-friendly LDS ----
    {
        int point = base + tid;
        int ray = point / NSTEPS, step = point % NSTEPS;
        float t = NEARV + DISTV * (float)step;
        float dx = rays_d[ray * 3 + 0], dy = rays_d[ray * 3 + 1], dz = rays_d[ray * 3 + 2];
        float px = rays_o[ray * 3 + 0] + dx * t;
        float py = rays_o[ray * 3 + 1] + dy * t;
        float pz = rays_o[ray * 3 + 2] + dz * t;

        int x0, y0, z0; float fx, fy, fz;
        tsetup(px, GG, x0, fx); tsetup(py, GG, y0, fy); tsetup(pz, GG, z0, fz);
        _Float16* f = feat + tid * FS;
#pragma unroll
        for (int c = 0; c < 3; ++c) {
            f[c] = (_Float16)trilerp_ch(offc + (long)c * G3, GG, x0, y0, z0, fx, fy, fz);
            emocol[tid * 4 + c] =
                (_Float16)trilerp_ch(emoc_g + (long)c * G3, GG, x0, y0, z0, fx, fy, fz);
        }
        float r3[3] = {(px + 1.f) * 0.5f, (py + 1.f) * 0.5f, (pz + 1.f) * 0.5f};
        f[3] = (_Float16)r3[0]; f[4] = (_Float16)r3[1]; f[5] = (_Float16)r3[2];
#pragma unroll
        for (int d = 0; d < 3; ++d) {
            float fr = 1.f;
#pragma unroll
            for (int q = 0; q < 5; ++q) {
                float ang = r3[d] * fr;
                f[6 + d * 5 + q]  = (_Float16)sinf(ang);
                f[21 + d * 5 + q] = (_Float16)cosf(ang);
                fr *= 2.f;
            }
        }
        float vv[3] = {viewdirs[ray * 3 + 0], viewdirs[ray * 3 + 1], viewdirs[ray * 3 + 2]};
#pragma unroll
        for (int d = 0; d < 3; ++d) {
            float fr = 1.f;
#pragma unroll
            for (int q = 0; q < 4; ++q) {
                float ang = vv[d] * fr;
                f[36 + d * 4 + q] = (_Float16)ang;
                f[48 + d * 4 + q] = (_Float16)sinf(ang);
                f[60 + d * 4 + q] = (_Float16)cosf(ang);
                fr *= 2.f;
            }
        }
        float g0 = gpt[point * 3 + 0], g1 = gpt[point * 3 + 1], g2 = gpt[point * 3 + 2];
        float inv = 1.0f / (sqrtf(g0 * g0 + g1 * g1 + g2 * g2) + 1e-5f);
        f[72] = (_Float16)(g0 * inv); f[73] = (_Float16)(g1 * inv); f[74] = (_Float16)(g2 * inv);
#pragma unroll
        for (int q = 75; q < FS; ++q) f[q] = (_Float16)0.f;
        emf[tid] = em_modes[ray];
    }
    __syncthreads();

    // ---- phase 2: run both MLPs on the matrix cores ----
    for (int m = 0; m < 2; ++m) {
        if (m == 1) {
#pragma unroll
            for (int c = 0; c < 3; ++c) feat[tid * FS + c] = emocol[tid * 4 + c];
            __syncthreads();
        }
        const _Float16* wp = wpack + m * WP_MLP;
        const float* b0 = m ? emo_b0 : off_b0;
        const float* b1 = m ? emo_b1 : off_b1;
        const float* b2 = m ? emo_b2 : off_b2;

        wmma_layer(feat, FS, 3, wp + WP_L0, 8, b0, hA, wave, lane);   // 75(->96) -> 128
        __syncthreads();
        wmma_layer(hA, 128, 4, wp + WP_L1, 8, b1, hB, wave, lane);    // 128 -> 128
        __syncthreads();

        // layer 2: 128 -> 3 (N padded to 16)
        for (int mt = wave * 2; mt < wave * 2 + 2; ++mt) {
            int row = mt * 16 + (lane & 15);
            int kb  = (lane & 16) ? 8 : 0;
            float bn = (lane & 15) < 3 ? b2[lane & 15] : 0.f;
            v8f c = {bn, bn, bn, bn, bn, bn, bn, bn};
            for (int kt = 0; kt < 4; ++kt) {
                v16h a = load_a_frag(hB + row * 128 + kt * 32 + kb);
                v16h b = *reinterpret_cast<const v16h*>(wp + WP_L2 + (kt * 32 + lane) * 16);
                c = __builtin_amdgcn_wmma_f32_16x16x32_f16(false, a, false, b,
                                                           (short)0, c, false, false);
            }
            int n = lane & 15;
            if (n < 3) {
                int mbase = mt * 16 + ((lane >> 4) << 3);
#pragma unroll
                for (int r = 0; r < 8; ++r) {
                    float v = sigmoidf(c[r]);
                    int rp = mbase + r;
                    if (m == 0)           outrgb[rp * 3 + n]  = v;
                    else if (emf[rp])     outrgb[rp * 3 + n] += v;
                }
            }
        }
        __syncthreads();
    }

    // ---- phase 3: weight and emit per-point rgb ----
    {
        int point = base + tid;
        float w = weights[point];
#pragma unroll
        for (int c = 0; c < 3; ++c) wrgb[point * 3 + c] = w * outrgb[tid * 3 + c];
    }
}

__global__ void reduce_kernel(const float* __restrict__ wrgb, float* __restrict__ out) {
    int ray = blockIdx.x * blockDim.x + threadIdx.x;
    if (ray >= NRAYS) return;
    float s0 = 0.f, s1 = 0.f, s2 = 0.f;
    for (int s = 0; s < NSTEPS; ++s) {
        int p = ray * NSTEPS + s;
        s0 += wrgb[p * 3 + 0]; s1 += wrgb[p * 3 + 1]; s2 += wrgb[p * 3 + 2];
    }
    out[ray * 3 + 0] = s0; out[ray * 3 + 1] = s1; out[ray * 3 + 2] = s2;
}

// ---------------- host launch ----------------

extern "C" void kernel_launch(void* const* d_in, const int* in_sizes, int n_in,
                              void* d_out, int out_size, void* d_ws, size_t ws_size,
                              hipStream_t stream) {
    const float* rays_o   = (const float*)d_in[0];
    const float* rays_d   = (const float*)d_in[1];
    const float* viewdirs = (const float*)d_in[2];
    const int*   em_modes = (const int*)d_in[3];
    const float* sdf_grid = (const float*)d_in[4];
    const float* mask_density = (const float*)d_in[5];
    const float* off_color = (const float*)d_in[6];
    const float* emo_color = (const float*)d_in[7];
    const float* off_w0 = (const float*)d_in[8],  *off_b0 = (const float*)d_in[9];
    const float* off_w1 = (const float*)d_in[10], *off_b1 = (const float*)d_in[11];
    const float* off_w2 = (const float*)d_in[12], *off_b2 = (const float*)d_in[13];
    const float* emo_w0 = (const float*)d_in[14], *emo_b0 = (const float*)d_in[15];
    const float* emo_w1 = (const float*)d_in[16], *emo_b1 = (const float*)d_in[17];
    const float* emo_w2 = (const float*)d_in[18], *emo_b2 = (const float*)d_in[19];

    float* ws   = (float*)d_ws;
    float* sdfs = ws;                          // G3 (final smoothed SDF)
    float* tmp  = ws + (size_t)G3;             // G3 scratch; reused as grad_pt
    float* grad = ws + (size_t)2 * G3;         // 3*G3
    float* alpha   = ws + (size_t)5 * G3;      // NPTS
    float* weights = alpha + NPTS;             // NPTS
    float* wrgb    = weights + NPTS;           // NPTS*3
    _Float16* wpack = (_Float16*)(wrgb + (size_t)NPTS * 3);  // 61440 halves
    float* gpt = tmp;                          // NPTS*3 (fits in G3), free after conv
    float* out = (float*)d_out;

    const int tb = 256;
    const int gb = (G3 + tb - 1) / tb;
    gauss1d<<<gb, tb, 0, stream>>>(sdf_grid, sdfs, GG * GG, 0);  // x: in -> A
    gauss1d<<<gb, tb, 0, stream>>>(sdfs, tmp, GG, 1);            // y: A -> B
    gauss1d<<<gb, tb, 0, stream>>>(tmp, sdfs, 1, 2);             // z: B -> A
    grad_kernel<<<gb, tb, 0, stream>>>(sdfs, grad);
    pack_weights<<<(2 * WP_MLP + tb - 1) / tb, tb, 0, stream>>>(
        off_w0, off_w1, off_w2, emo_w0, emo_w1, emo_w2, wpack);
    alpha_kernel<<<(NPTS + tb - 1) / tb, tb, 0, stream>>>(
        rays_o, rays_d, viewdirs, mask_density, sdfs, grad, alpha, gpt);
    weights_kernel<<<(NRAYS + tb - 1) / tb, tb, 0, stream>>>(alpha, weights, out);
    mlp_kernel<<<NPTS / PTS, PTS, 0, stream>>>(
        rays_o, rays_d, viewdirs, em_modes, off_color, emo_color, wpack,
        off_b0, off_b1, off_b2, emo_b0, emo_b1, emo_b2, gpt, weights, wrgb);
    reduce_kernel<<<(NRAYS + tb - 1) / tb, tb, 0, stream>>>(wrgb, out);
}
